// NeuralODE_5257039970809
// MI455X (gfx1250) — compile-verified, matrix operand order checked by reading, not run on hardware
//
#include <hip/hip_runtime.h>
#include <math.h>

// ---------------------------------------------------------------------------
// Euler-Maruyama SDE scan with tanh-MLP drift, mapped to CDNA5 WMMA (bf16).
//   B=4096, D=64, W=256, T=256.  out: [257][4096][64] f32.
// Each workgroup owns 32 batch rows (128 persistent WGs -> fills >=128 WGPs)
// and runs the entire 256-step scan locally; no grid-level sync needed.
// Hidden weights (2 x 256x256 bf16 = 256KB) live in LDS for the whole kernel.
// ---------------------------------------------------------------------------

typedef __attribute__((ext_vector_type(16))) __bf16 v16bf;
typedef __attribute__((ext_vector_type(8)))  float  v8f;

#define B_TOT  4096
#define D_DIM  64
#define W_DIM  256
#define NSTEP  256
#define M_TILE 32
#define SIGMA  0.1f

union FragBF { uint4 u[2]; v16bf v; };

// Native CDNA5 V_TANH_F32 (confirmed lowering in round 2).
__device__ __forceinline__ float fast_tanh(float x) {
#if __has_builtin(__builtin_amdgcn_tanhf)
  return __builtin_amdgcn_tanhf(x);
#elif __has_builtin(__builtin_amdgcn_tanh_f32)
  return __builtin_amdgcn_tanh_f32(x);
#else
  return tanhf(x);
#endif
}

// Load one 16x32 bf16 WMMA A/B fragment from a row-major [rows][rowStride]
// array. Per ISA 7.12.2 (16-bit A-matrix 16x32): lanes 0-15 hold K 0..7 in
// VGPR0-3 and K 16..23 in VGPR4-7; lanes 16-31 hold K 8..15 and K 24..31.
__device__ __forceinline__ v16bf load_frag(const __bf16* base, int rowStride,
                                           int row0, int k0, int lane) {
  const int half = lane >> 4;      // 0 or 1
  const int r    = lane & 15;      // row (A) / column (B) within tile
  const __bf16* p = base + (size_t)(row0 + r) * rowStride + k0 + half * 8;
  FragBF f;
  f.u[0] = *(const uint4*)(p);       // K = +0..7   (or +8..15)
  f.u[1] = *(const uint4*)(p + 16);  // K = +16..23 (or +24..31)
  return f.v;
}

// One dense layer: outLds[32][N] = tanh( in[32][K] @ wB[N][K]^T + bias ).
// Wave `wave` owns output columns [32*wave, 32*wave+32) (N == 256).
// K-loop fragments are double-buffered so LDS latency hides behind WMMAs.
template <int K>
__device__ __forceinline__ void dense_tanh_layer(
    const __bf16* __restrict__ in, int inStride,
    const __bf16* __restrict__ wB, int wStride,
    const float* __restrict__ bias,
    __bf16* __restrict__ outLds, int outStride,
    int wave, int lane) {
  constexpr int KB = K >> 5;
  const int n0 = wave * 32;
  v8f acc[2][2];
  v8f zero = {0.f, 0.f, 0.f, 0.f, 0.f, 0.f, 0.f, 0.f};
#pragma unroll
  for (int m = 0; m < 2; ++m) { acc[m][0] = zero; acc[m][1] = zero; }

  v16bf bf[2][2], af[2][2];
  bf[0][0] = load_frag(wB, wStride, n0,      0, lane);
  bf[0][1] = load_frag(wB, wStride, n0 + 16, 0, lane);
  af[0][0] = load_frag(in, inStride, 0,  0, lane);
  af[0][1] = load_frag(in, inStride, 16, 0, lane);

#pragma unroll
  for (int kb = 0; kb < KB; ++kb) {
    const int cur = kb & 1, nxt = cur ^ 1;
    if (kb + 1 < KB) {  // prefetch next K-slice while this one computes
      const int k0 = (kb + 1) * 32;
      bf[nxt][0] = load_frag(wB, wStride, n0,      k0, lane);
      bf[nxt][1] = load_frag(wB, wStride, n0 + 16, k0, lane);
      af[nxt][0] = load_frag(in, inStride, 0,  k0, lane);
      af[nxt][1] = load_frag(in, inStride, 16, k0, lane);
    }
#pragma unroll
    for (int m = 0; m < 2; ++m) {
      acc[m][0] = __builtin_amdgcn_wmma_f32_16x16x32_bf16(
          false, af[cur][m], false, bf[cur][0], (short)0, acc[m][0],
          false, false);
      acc[m][1] = __builtin_amdgcn_wmma_f32_16x16x32_bf16(
          false, af[cur][m], false, bf[cur][1], (short)0, acc[m][1],
          false, false);
    }
  }

  // C/D layout: lane (0-15|16-31) holds column n0+t*16+(lane&15),
  // rows m*16 + (half?8:0) + r for VGPR r.  Batch the tanh evaluations per
  // tile (8 independent TRANS chains) before converting/storing.
  const int cn = lane & 15, half = lane >> 4;
#pragma unroll
  for (int m = 0; m < 2; ++m) {
#pragma unroll
    for (int t = 0; t < 2; ++t) {
      const int n  = n0 + t * 16 + cn;
      const float bv = bias[n];
      const int r0 = m * 16 + half * 8;
      float tv[8];
#pragma unroll
      for (int r = 0; r < 8; ++r) tv[r] = fast_tanh(acc[m][t][r] + bv);
#pragma unroll
      for (int r = 0; r < 8; ++r)
        outLds[(size_t)(r0 + r) * outStride + n] = (__bf16)tv[r];
    }
  }
}

__global__ void __launch_bounds__(256)
sde_scan_kernel(const float* __restrict__ ts, const float* __restrict__ y0,
                const float* __restrict__ dW, const float* __restrict__ b_in,
                const float* __restrict__ b_h, const float* __restrict__ b_out,
                const __bf16* __restrict__ w_in_bf,
                const __bf16* __restrict__ wh_bf,   // [2][256][256]
                const __bf16* __restrict__ w_out_bf,
                float* __restrict__ out) {
  extern __shared__ __align__(16) char smem_raw[];
  __bf16* s_a   = (__bf16*)smem_raw;          // 32*256 bf16 (16KB) ping
  __bf16* s_b   = s_a + M_TILE * 256;         // 32*256 bf16 (16KB) pong
  __bf16* s_wh0 = s_b + M_TILE * 256;         // 256*256 bf16 (128KB)
  // s_wh1 = s_wh0 + 65536                      256*256 bf16 (128KB) => 288KB

  // ---- Preload both hidden weight matrices into LDS (once per kernel) ----
  {
    const uint4* src = (const uint4*)wh_bf;   // 2*256*256 bf16 = 16384 uint4
    uint4* dst = (uint4*)s_wh0;
    for (int i = threadIdx.x; i < 16384; i += 256) dst[i] = src[i];
  }

  const int lane  = threadIdx.x & 31;
  const int wave  = threadIdx.x >> 5;   // 0..7
  const int cn    = lane & 15;
  const int half  = lane >> 4;
  const int rowWG = blockIdx.x * M_TILE;

  // Output-layer tile ownership: 8 tiles (2m x 4n), exactly one per wave.
  const int ntile = wave >> 1;          // 0..3
  const int mtile = wave & 1;           // 0..1
  const int ycol  = ntile * 16 + cn;
  const int ym0   = mtile * 16 + half * 8;   // local row base for this lane

  // ---- y in registers, in the output-layer C-tile layout; emit out[0] ----
  float y_reg[8];
#pragma unroll
  for (int r = 0; r < 8; ++r) {
    const size_t row = (size_t)(rowWG + ym0 + r);
    const float v = y0[row * D_DIM + ycol];
    y_reg[r] = v;
    out[row * D_DIM + ycol] = v;                       // out[0] = y0
    s_a[(size_t)(ym0 + r) * D_DIM + ycol] = (__bf16)v; // ybf, stride 64
  }
  __syncthreads();

  // ------------------------------ time scan -------------------------------
  for (int t = 0; t < NSTEP; ++t) {
    // Prefetch next step's Brownian increments (hides HBM latency behind
    // ~3 GEMM layers of compute); lowers to global_prefetch_b8.
    if (t + 1 < NSTEP) {
      const float* dWn = dW + (size_t)(t + 1) * B_TOT * D_DIM;
#pragma unroll
      for (int r = 0; r < 8; ++r) {
        __builtin_prefetch(dWn + (size_t)(rowWG + ym0 + r) * D_DIM + ycol,
                           0, 3);
      }
    }

    // L1: h1 = tanh(y @ w_in^T)          [32x64] x [64x256]
    dense_tanh_layer<D_DIM>(s_a, D_DIM, w_in_bf, D_DIM, b_in, s_b, W_DIM,
                            wave, lane);
    __syncthreads();
    // L2: h2 = tanh(h1 @ w_h[0]^T)       [32x256] x [256x256]   (LDS weights)
    dense_tanh_layer<W_DIM>(s_b, W_DIM, s_wh0, W_DIM, b_h, s_a, W_DIM,
                            wave, lane);
    __syncthreads();
    // L3: h3 = tanh(h2 @ w_h[1]^T)
    dense_tanh_layer<W_DIM>(s_a, W_DIM, s_wh0 + 256 * 256, W_DIM,
                            b_h + W_DIM, s_b, W_DIM, wave, lane);
    __syncthreads();

    // L4: o = h3 @ w_out^T + b_out; y += dt*(-y+o) + sigma*sqrt(dt)*dW[t]
    v8f accO = {0.f, 0.f, 0.f, 0.f, 0.f, 0.f, 0.f, 0.f};
    {
      v16bf bb[2], aa[2];
      bb[0] = load_frag(w_out_bf, W_DIM, ntile * 16, 0, lane);
      aa[0] = load_frag(s_b, W_DIM, mtile * 16, 0, lane);
#pragma unroll
      for (int kb = 0; kb < 8; ++kb) {
        const int cur = kb & 1, nxt = cur ^ 1;
        if (kb + 1 < 8) {
          const int k0 = (kb + 1) * 32;
          bb[nxt] = load_frag(w_out_bf, W_DIM, ntile * 16, k0, lane);
          aa[nxt] = load_frag(s_b, W_DIM, mtile * 16, k0, lane);
        }
        accO = __builtin_amdgcn_wmma_f32_16x16x32_bf16(
            false, aa[cur], false, bb[cur], (short)0, accO, false, false);
      }
    }

    const float dt = ts[t + 1] - ts[t];
    const float sq = SIGMA * __builtin_sqrtf(dt);
    const float bo = b_out[ycol];
    const float* dWt  = dW  + (size_t)t * B_TOT * D_DIM;
    float*       outT = out + (size_t)(t + 1) * B_TOT * D_DIM;

    // Batch: load noise, compute all updates, then store (independent chains)
    float dwv[8];
#pragma unroll
    for (int r = 0; r < 8; ++r)
      dwv[r] = dWt[(size_t)(rowWG + ym0 + r) * D_DIM + ycol];
    float yn[8];
#pragma unroll
    for (int r = 0; r < 8; ++r) {
      const float yv = y_reg[r];
      yn[r] = yv + dt * (-yv + (accO[r] + bo)) + sq * dwv[r];
      y_reg[r] = yn[r];
    }
#pragma unroll
    for (int r = 0; r < 8; ++r) {
      const size_t row = (size_t)(rowWG + ym0 + r);
      outT[row * D_DIM + ycol] = yn[r];
      s_a[(size_t)(ym0 + r) * D_DIM + ycol] = (__bf16)yn[r];  // next-step ybf
    }
    __syncthreads();
  }
}

// One-shot fp32 -> bf16 weight conversion into d_ws (keeps native
// [out_dim][in_dim] layout, which is exactly the WMMA B-fragment layout).
__global__ void __launch_bounds__(256)
prep_weights_kernel(const float* __restrict__ w_in,
                    const float* __restrict__ w_h,
                    const float* __restrict__ w_out, __bf16* __restrict__ ws) {
  const int i = blockIdx.x * blockDim.x + threadIdx.x;  // 0 .. 131071
  if (i < 16384) ws[i] = (__bf16)w_in[i];                      // [256][64]
  ws[16384 + i] = (__bf16)w_h[i];                              // [2][256][256]
  if (i < 16384) ws[16384 + 131072 + i] = (__bf16)w_out[i];    // [64][256]
}

extern "C" void kernel_launch(void* const* d_in, const int* in_sizes, int n_in,
                              void* d_out, int out_size, void* d_ws,
                              size_t ws_size, hipStream_t stream) {
  const float* ts    = (const float*)d_in[0];
  const float* y0    = (const float*)d_in[1];
  const float* dW    = (const float*)d_in[2];
  const float* w_in  = (const float*)d_in[3];
  const float* b_in  = (const float*)d_in[4];
  const float* w_h   = (const float*)d_in[5];
  const float* b_h   = (const float*)d_in[6];
  const float* w_out = (const float*)d_in[7];
  const float* b_out = (const float*)d_in[8];
  float* out = (float*)d_out;

  __bf16* wbf = (__bf16*)d_ws;
  prep_weights_kernel<<<512, 256, 0, stream>>>(w_in, w_h, w_out, wbf);

  const __bf16* w_in_bf  = wbf;
  const __bf16* wh_bf    = wbf + 16384;
  const __bf16* w_out_bf = wbf + 16384 + 131072;

  // LDS: 2 activation buffers (2*16KB) + 2 hidden weight mats (2*128KB) = 288KB
  const size_t shmem =
      (size_t)(2 * M_TILE * 256 + 2 * 256 * 256) * sizeof(__bf16);
  sde_scan_kernel<<<B_TOT / M_TILE, 256, shmem, stream>>>(
      ts, y0, dW, b_in, b_h, b_out, w_in_bf, wh_bf, w_out_bf, out);
}
